// RealizedGARCHPINN_55087250538689
// MI455X (gfx1250) — matrix operand order repaired and checked by reading, not compiled, not run on hardware
//
#include <hip/hip_runtime.h>

// ---------------------------------------------------------------------------
// RealizedGARCH-PINN scan for MI455X (gfx1250, wave32).
//
// Grid: B/32 = 128 blocks of one wave32. Each wave owns 32 batch rows and
// walks all T=2048 steps (the scan is inherently sequential in T, so max
// parallelism = B lanes; optimize per-step latency, not throughput).
// Per step: GARCH recurrence (VALU) -> measurement eq (VALU) ->
// MLP 3->64->64->1 via v_wmma_f32_16x16x32_f16 (layers 1&2) + VALU dot
// (layer 3). Weights live in registers for the whole kernel; one LDS
// round-trip per step re-lays layer-1 D-fragments into layer-2 A-fragments.
// ReLU uses fmaxf so the backend's hazard recognizer provably inserts the
// mandatory WMMA->VALU nops (inline-asm reads were shown to skip them).
// ---------------------------------------------------------------------------

typedef __attribute__((ext_vector_type(16))) _Float16 v16h;
typedef __attribute__((ext_vector_type(8)))  float    v8f;

#define GB 4096
#define GT 2048
#define GH 64
#define LSTR 68   // f32 stride of the 32x64 activation tile in LDS (pad vs banks)

__device__ __forceinline__ v16h vzero16h() {
    v16h v;
#pragma unroll
    for (int i = 0; i < 16; ++i) v[i] = (_Float16)0.0f;
    return v;
}

__device__ __forceinline__ v8f splat8(float x) {
    v8f v;
#pragma unroll
    for (int i = 0; i < 8; ++i) v[i] = x;
    return v;
}

__global__ __launch_bounds__(32)
void garch_mlp_scan(const float* __restrict__ g_ret,   // [B,T]
                    const float* __restrict__ g_lrv,   // [B,T]
                    const float* __restrict__ p_omega, const float* __restrict__ p_beta,
                    const float* __restrict__ p_tau1,  const float* __restrict__ p_tau2,
                    const float* __restrict__ p_gamma, const float* __restrict__ p_xi,
                    const float* __restrict__ p_phi,   const float* __restrict__ p_d1,
                    const float* __restrict__ p_d2,    const float* __restrict__ p_mu,
                    const float* __restrict__ W1, const float* __restrict__ b1,
                    const float* __restrict__ W2, const float* __restrict__ b2,
                    const float* __restrict__ W3, const float* __restrict__ b3,
                    float* __restrict__ o_lh, float* __restrict__ o_lx,
                    float* __restrict__ o_z,  float* __restrict__ o_u)
{
    const int  lane = threadIdx.x & 31;
    const int  row  = blockIdx.x * 32 + lane;
    const bool hi   = lane >= 16;
    const int  nlo  = lane & 15;

    // ---- scalar GARCH parameters (uniform) --------------------------------
    const float omega = p_omega[0], beta = p_beta[0];
    const float tau1  = p_tau1[0],  tau2 = p_tau2[0], gamma = p_gamma[0];
    const float xi    = p_xi[0],    phi  = p_phi[0];
    const float d1    = p_d1[0],    d2   = p_d2[0],   mu    = p_mu[0];

    // ---- W1 as f16 B-fragments (K=3 padded to 32), 4 N-tiles --------------
    // B operand in this code's contiguous K order (32x16): lanes 0-15 hold col=lane&15, K = element e;
    // lanes 16-31 hold K = 16+e -> all zero here.
    v16h w1f[4];
#pragma unroll
    for (int nt = 0; nt < 4; ++nt) {
        v16h f = vzero16h();
        if (!hi) {
            const int c = nt * 16 + nlo;
            f[0] = (_Float16)W1[0 * GH + c];
            f[1] = (_Float16)W1[1 * GH + c];
            f[2] = (_Float16)W1[2 * GH + c];
        }
        w1f[nt] = f;
    }

    // ---- W2 as f16 B-fragments: 2 K-chunks x 4 N-tiles --------------------
    // lanes 0-15: col = nt*16+lane, K = kc*32 + e; lanes 16-31: K += 16.
    v16h w2f[2][4];
#pragma unroll
    for (int kc = 0; kc < 2; ++kc)
#pragma unroll
        for (int nt = 0; nt < 4; ++nt) {
            const int c  = nt * 16 + nlo;
            const int kb = kc * 32 + (hi ? 16 : 0);
            v16h f;
#pragma unroll
            for (int e = 0; e < 16; ++e)
                f[e] = (_Float16)W2[(kb + e) * GH + c];
            w2f[kc][nt] = f;
        }

    // ---- biases + W3 in per-lane registers (D-layout aligned) -------------
    float b1c[4], b2c[4], w3v[4][8];
#pragma unroll
    for (int nt = 0; nt < 4; ++nt) {
        b1c[nt] = b1[nt * 16 + nlo];
        b2c[nt] = b2[nt * 16 + nlo];
#pragma unroll
        for (int i = 0; i < 8; ++i)
            w3v[nt][i] = W3[nt * 16 + i + (hi ? 8 : 0)];
    }
    const float b3s = b3[0];

    // ---- LDS staging tile for layer1 -> layer2 re-layout ------------------
    __shared__ float sh1[32 * LSTR];

    const long rowT = (long)row * GT;
    float lh = 0.0f, zp = 0.0f, up = 0.0f;          // carry
    float r_cur = g_ret[rowT], lv_cur = g_lrv[rowT];

#pragma unroll 1
    for (int t = 0; t < GT; ++t) {
        // software-pipelined operand prefetch (branchless clamped indices)
        const int tn = (t + 1  < GT) ? t + 1  : GT - 1;   // s_min, no branch
        const int tp = (t + 32 < GT) ? t + 32 : GT - 1;
        const float r_nxt  = g_ret[rowT + tn];
        const float lv_nxt = g_lrv[rowT + tn];
        __builtin_prefetch(&g_ret[rowT + tp], 0, 0);      // global_prefetch_b8
        __builtin_prefetch(&g_lrv[rowT + tp], 0, 0);

        // ---- recurrence + measurement -------------------------------------
        float log_h;
        if (t == 0) log_h = lv_cur;   // log(mean(exp(lrv[0:1]))) == lrv[:,0]
        else        log_h = omega + beta * lh + tau1 * zp
                          + tau2 * (zp * zp - 1.0f) + gamma * up;

        const float hvar = __expf(log_h);
        const float zt   = (r_cur - mu) * rsqrtf(hvar);
        const float dlt  = d1 * zt + d2 * (zt * zt - 1.0f);
        const float sfit = xi + phi * log_h + dlt;
        const float ut   = lv_cur - sfit;
        const float lxt  = sfit + ut;

        // ---- layer 1: x[32,3] @ W1[3,64]  (8x wmma f16, K padded) ---------
        const float lh_sw = __shfl_xor(log_h, 16, 32);
        const float z_sw  = __shfl_xor(zt,    16, 32);
        const float u_sw  = __shfl_xor(ut,    16, 32);

        v16h a1t[2];
        {
            v16h a0 = vzero16h(), a1 = vzero16h();
            if (!hi) {   // A layout: lanes 0-15 hold row=lane, elements 0..7 = K0..7
                a0[0] = (_Float16)log_h; a0[1] = (_Float16)zt;   a0[2] = (_Float16)ut;
                a1[0] = (_Float16)lh_sw; a1[1] = (_Float16)z_sw; a1[2] = (_Float16)u_sw;
            }
            a1t[0] = a0; a1t[1] = a1;
        }

#pragma unroll
        for (int mt = 0; mt < 2; ++mt)
#pragma unroll
            for (int nt = 0; nt < 4; ++nt) {
                v8f c = splat8(b1c[nt]);
                c = __builtin_amdgcn_wmma_f32_16x16x32_f16(
                        false, a1t[mt], false, w1f[nt], (short)0, c, false, false);
                // ReLU (v_max_num_f32, hazard-nops inserted by backend)
                // + scatter to LDS row-major [32][LSTR]
                const int ccol  = nt * 16 + nlo;
                const int rbase = mt * 16 + (hi ? 8 : 0);
#pragma unroll
                for (int i = 0; i < 8; ++i)
                    sh1[(rbase + i) * LSTR + ccol] = fmaxf(c[i], 0.0f);
            }

        asm volatile("s_wait_dscnt 0" ::: "memory");   // LDS store->load turn

        // ---- layer 2: h1[32,64] @ W2[64,64]  (16x wmma f16) ---------------
        v8f acc[2][4];
#pragma unroll
        for (int mt = 0; mt < 2; ++mt)
#pragma unroll
            for (int nt = 0; nt < 4; ++nt)
                acc[mt][nt] = splat8(b2c[nt]);

#pragma unroll
        for (int mt = 0; mt < 2; ++mt) {
            const int arow = mt * 16 + nlo;
#pragma unroll
            for (int kc = 0; kc < 2; ++kc) {
                const int kb = kc * 32 + (hi ? 16 : 0);
                const float4 q0 = *(const float4*)&sh1[arow * LSTR + kb];
                const float4 q1 = *(const float4*)&sh1[arow * LSTR + kb + 4];
                const float4 q2 = *(const float4*)&sh1[arow * LSTR + kb + 8];
                const float4 q3 = *(const float4*)&sh1[arow * LSTR + kb + 12];
                v16h a;
                a[0]  = (_Float16)q0.x; a[1]  = (_Float16)q0.y;
                a[2]  = (_Float16)q0.z; a[3]  = (_Float16)q0.w;
                a[4]  = (_Float16)q1.x; a[5]  = (_Float16)q1.y;
                a[6]  = (_Float16)q1.z; a[7]  = (_Float16)q1.w;
                a[8]  = (_Float16)q2.x; a[9]  = (_Float16)q2.y;
                a[10] = (_Float16)q2.z; a[11] = (_Float16)q2.w;
                a[12] = (_Float16)q3.x; a[13] = (_Float16)q3.y;
                a[14] = (_Float16)q3.z; a[15] = (_Float16)q3.w;
#pragma unroll
                for (int nt = 0; nt < 4; ++nt)
                    acc[mt][nt] = __builtin_amdgcn_wmma_f32_16x16x32_f16(
                        false, a, false, w2f[kc][nt], (short)0, acc[mt][nt],
                        false, false);
            }
        }

        // ---- layer 3: ReLU + dot with W3 (VALU, N=1) ----------------------
        float s3[2] = {0.0f, 0.0f};
#pragma unroll
        for (int mt = 0; mt < 2; ++mt)
#pragma unroll
            for (int nt = 0; nt < 4; ++nt)
#pragma unroll
                for (int i = 0; i < 8; ++i)
                    s3[mt] += fmaxf(acc[mt][nt][i], 0.0f) * w3v[nt][i];

        s3[0] += __shfl_xor(s3[0], 16, 32);   // combine j-halves across lane halves
        s3[1] += __shfl_xor(s3[1], 16, 32);

        const float nn  = b3s + (hi ? s3[1] : s3[0]);
        const float enh = log_h + 0.01f * nn;

        // ---- emit + carry -------------------------------------------------
        const long o = rowT + t;
        o_lh[o] = enh; o_lx[o] = lxt; o_z[o] = zt; o_u[o] = ut;

        lh = enh; zp = zt; up = ut;
        r_cur = r_nxt; lv_cur = lv_nxt;
    }
}

extern "C" void kernel_launch(void* const* d_in, const int* in_sizes, int n_in,
                              void* d_out, int out_size, void* d_ws, size_t ws_size,
                              hipStream_t stream) {
    (void)in_sizes; (void)n_in; (void)out_size; (void)d_ws; (void)ws_size;
    const float* g_ret = (const float*)d_in[0];
    const float* g_lrv = (const float*)d_in[1];
    const float* om = (const float*)d_in[2];
    const float* be = (const float*)d_in[3];
    const float* t1 = (const float*)d_in[4];
    const float* t2 = (const float*)d_in[5];
    const float* ga = (const float*)d_in[6];
    const float* xi = (const float*)d_in[7];
    const float* ph = (const float*)d_in[8];
    const float* d1 = (const float*)d_in[9];
    const float* d2 = (const float*)d_in[10];
    const float* mu = (const float*)d_in[11];
    const float* W1 = (const float*)d_in[12];
    const float* b1 = (const float*)d_in[13];
    const float* W2 = (const float*)d_in[14];
    const float* b2 = (const float*)d_in[15];
    const float* W3 = (const float*)d_in[16];
    const float* b3 = (const float*)d_in[17];

    float* out = (float*)d_out;
    const long BT = (long)GB * GT;

    garch_mlp_scan<<<GB / 32, 32, 0, stream>>>(
        g_ret, g_lrv, om, be, t1, t2, ga, xi, ph, d1, d2, mu,
        W1, b1, W2, b2, W3, b3,
        out, out + BT, out + 2 * BT, out + 3 * BT);
}
